// MultiheadSelfAttention_15315853377651
// MI455X (gfx1250) — matrix operand order repaired and checked by reading, not compiled, not run on hardware
//
#include <hip/hip_runtime.h>
#include <hip/hip_bf16.h>

typedef unsigned short u16;
typedef unsigned int   u32;

typedef __attribute__((ext_vector_type(16))) __bf16 v16bf;
typedef __attribute__((ext_vector_type(8)))  float  v8f;

union ABfrag { u32 u[8]; v16bf v; };

__device__ inline u16 f2b(float f) {
  u32 u = __float_as_uint(f);
  u32 r = (u + 0x7FFFu + ((u >> 16) & 1u)) >> 16;   // RNE f32 -> bf16
  return (u16)r;
}
__device__ inline float b2f(u16 h) { return __uint_as_float(((u32)h) << 16); }

__device__ inline v8f wmma_bf16(v16bf a, v16bf b, v8f c) {
  return __builtin_amdgcn_wmma_f32_16x16x32_bf16(false, a, false, b, (short)0, c,
                                                 false, false);
}

// Async DMA: global -> LDS, 16 bytes per lane, tracked by ASYNCcnt (GV mode).
__device__ inline void async_b128_to_lds(u32 lds_off, const void* gaddr) {
  asm volatile("global_load_async_to_lds_b128 %0, %1, off"
               :
               : "v"(lds_off), "v"(gaddr)
               : "memory");
}

// ---------------------------------------------------------------------------
// Tiled GEMM: C[8192x1024] = A[8192x1024] * W[1024x1024]
// AFMT: 0 = A is f32 (convert to bf16 in LDS), 1 = A is bf16 already
// EPI:  0 = scatter bf16 into [B,H,T,D] attention layout, 1 = plain f32 out
// Block = 256 threads (8 waves), BM=BN=128, BK=32 (one WMMA k-step per iter).
// Wave grid 4x2; each wave computes a 32x64 tile = 2x4 WMMA 16x16 tiles.
// ---------------------------------------------------------------------------
template <int AFMT, int EPI>
__global__ __launch_bounds__(256) void gemm_kernel(const void* __restrict__ Aptr,
                                                   const float* __restrict__ W,
                                                   void* __restrict__ Out) {
  __shared__ __align__(16) u16 As[128][32];   // [m][k]
  __shared__ __align__(16) u16 Bs[128][32];   // [n][k]  (transposed B)
  const int tid  = threadIdx.x;
  const int lane = tid & 31;
  const int wv   = tid >> 5;
  const int wr   = wv >> 1;                   // 0..3
  const int wc   = wv & 1;                    // 0..1
  const int rr   = lane & 15;
  const int half = lane >> 4;
  const int rowBase = blockIdx.y * 128;
  const int colBase = blockIdx.x * 128;

  v8f acc[2][4];
  for (int i = 0; i < 2; ++i)
    for (int j = 0; j < 4; ++j)
      for (int r = 0; r < 8; ++r) acc[i][j][r] = 0.0f;

  for (int k0 = 0; k0 < 1024; k0 += 32) {
    if (AFMT == 0) {
      const float* A = (const float*)Aptr;
#pragma unroll
      for (int i = 0; i < 4; ++i) {
        int c = tid + 256 * i;
        int row = c >> 3, co = (c & 7) * 4;
        float4 v = *(const float4*)(A + (size_t)(rowBase + row) * 1024 + k0 + co);
        As[row][co + 0] = f2b(v.x); As[row][co + 1] = f2b(v.y);
        As[row][co + 2] = f2b(v.z); As[row][co + 3] = f2b(v.w);
      }
    } else {
      const u16* A = (const u16*)Aptr;
#pragma unroll
      for (int i = 0; i < 2; ++i) {
        int c = tid + 256 * i;
        int row = c >> 2, co = (c & 3) * 8;
        *(uint4*)&As[row][co] =
            *(const uint4*)(A + (size_t)(rowBase + row) * 1024 + k0 + co);
      }
    }
#pragma unroll
    for (int i = 0; i < 4; ++i) {
      int c = tid + 256 * i;
      int kk = c >> 5, co = (c & 31) * 4;
      float4 v = *(const float4*)(W + (size_t)(k0 + kk) * 1024 + colBase + co);
      Bs[co + 0][kk] = f2b(v.x); Bs[co + 1][kk] = f2b(v.y);
      Bs[co + 2][kk] = f2b(v.z); Bs[co + 3][kk] = f2b(v.w);
    }
    __syncthreads();

    // A fragments: lane<16 holds K {0..7,16..23}, lane>=16 holds K {8..15,24..31}
    ABfrag af[2];
#pragma unroll
    for (int i = 0; i < 2; ++i) {
      const u16* p = &As[wr * 32 + i * 16 + rr][0];
#pragma unroll
      for (int j = 0; j < 4; ++j) {
        af[i].u[j]     = *(const u32*)(p + half * 8 + 2 * j);
        af[i].u[4 + j] = *(const u32*)(p + half * 8 + 16 + 2 * j);
      }
    }
    // B fragments: lane<16 holds K {0..7,16..23}, lane>=16 holds K {8..15,24..31}
    ABfrag bf[4];
#pragma unroll
    for (int j = 0; j < 4; ++j) {
      const u16* p = &Bs[wc * 64 + j * 16 + rr][0];
#pragma unroll
      for (int q = 0; q < 8; ++q) bf[j].u[q] = *(const u32*)(p + half * 8 + (q >> 2) * 16 + 2 * (q & 3));
    }
#pragma unroll
    for (int i = 0; i < 2; ++i)
#pragma unroll
      for (int j = 0; j < 4; ++j) acc[i][j] = wmma_bf16(af[i].v, bf[j].v, acc[i][j]);
    __syncthreads();
  }

#pragma unroll
  for (int i = 0; i < 2; ++i)
#pragma unroll
    for (int j = 0; j < 4; ++j)
#pragma unroll
      for (int r = 0; r < 8; ++r) {
        int grow = rowBase + wr * 32 + i * 16 + half * 8 + r;
        int gcol = colBase + wc * 64 + j * 16 + rr;
        float vv = acc[i][j][r];
        if (EPI == 0) {
          int b = grow >> 11, t = grow & 2047;     // row = b*2048 + t
          int h = gcol >> 7, d = gcol & 127;       // col = h*128 + d
          ((u16*)Out)[(((size_t)(b * 8 + h)) * 2048 + t) * 128 + d] = f2b(vv);
        } else {
          ((float*)Out)[(size_t)grow * 1024 + gcol] = vv;
        }
      }
}

// ---------------------------------------------------------------------------
// Fractional RoPE (first 64 dims rotated pairing d <-> d+32) + D^-1/2 scale,
// in-place on bf16 Q and K laid out [B,H,T,D].
// ---------------------------------------------------------------------------
__global__ __launch_bounds__(256) void rope_kernel(u16* __restrict__ Q,
                                                   u16* __restrict__ K) {
  int idx = blockIdx.x * 256 + threadIdx.x;
  const int total = 4 * 8 * 2048 * 96;
  if (idx >= total) return;
  int i   = idx % 96;
  int bht = idx / 96;
  int t   = bht & 2047;
  int bh  = bht >> 11;
  const float scale = 0.088388347648318447f;      // 128^-0.5
  u16* qp = Q + ((size_t)bh * 2048 + t) * 128;
  u16* kp = K + ((size_t)bh * 2048 + t) * 128;
  if (i < 32) {
    float ang = __powf(10000.0f, -(float)i * (1.0f / 32.0f));
    float rad = (float)t * ang;
    float c = __cosf(rad), s = __sinf(rad);
    float qe = b2f(qp[i]), qo = b2f(qp[i + 32]);
    qp[i]      = f2b((qe * c - qo * s) * scale);
    qp[i + 32] = f2b((qe * s + qo * c) * scale);
    float ke = b2f(kp[i]), ko = b2f(kp[i + 32]);
    kp[i]      = f2b((ke * c - ko * s) * scale);
    kp[i + 32] = f2b((ke * s + ko * c) * scale);
  } else {
    int d = i + 32;                               // 64..127: pass-through half
    qp[d] = f2b(b2f(qp[d]) * scale);
    kp[d] = f2b(b2f(kp[d]) * scale);
  }
}

// ---------------------------------------------------------------------------
// Causal flash attention. Grid (T/128, B*H), 256 threads = 8 waves.
// Each wave owns 16 query rows (Q kept as A-fragments in registers).
// K/V key-blocks of 64 are DMA'd global->LDS with global_load_async_to_lds
// into DOUBLE-BUFFERED tiles; block kb+1 is prefetched while WMMAs for
// block kb run, synchronized with s_wait_asynccnt (async loads complete
// in order per wave). Online softmax per wave; P is round-tripped through
// a per-wave LDS buffer to convert C-layout -> A-layout.
// Output written bf16 to [B,T,H*D] for the final projection GEMM.
// ---------------------------------------------------------------------------
__global__ __launch_bounds__(256) void flash_kernel(const u16* __restrict__ Q,
                                                    const u16* __restrict__ K,
                                                    const u16* __restrict__ V,
                                                    u16* __restrict__ O) {
  __shared__ __align__(16) u16 Ks[2][64][128];  // [buf][key][d]
  __shared__ __align__(16) u16 Vs[2][64][128];  // [buf][key][d] (row major)
  __shared__ __align__(16) u16 Ps[8][16][64];   // per-wave P staging
  const int tid = threadIdx.x, lane = tid & 31, wv = tid >> 5;
  const int rr = lane & 15, half = lane >> 4;
  const int bh = blockIdx.y;
  const int qbase = blockIdx.x * 128;
  const u16* Qb = Q + (size_t)bh * 2048 * 128;
  const u16* Kb = K + (size_t)bh * 2048 * 128;
  const u16* Vb = V + (size_t)bh * 2048 * 128;

  // 8 async b128 ops per thread per key block (4 for K, 4 for V).
  auto prefetch = [&](int kb, int buf) {
#pragma unroll
    for (int i2 = 0; i2 < 4; ++i2) {
      int c = tid + 256 * i2;
      int jj = c >> 4, doff = (c & 15) * 8;
      u32 koff = (u32)(uintptr_t)&Ks[buf][jj][doff];
      u32 voff = (u32)(uintptr_t)&Vs[buf][jj][doff];
      async_b128_to_lds(koff, Kb + (size_t)(kb * 64 + jj) * 128 + doff);
      async_b128_to_lds(voff, Vb + (size_t)(kb * 64 + jj) * 128 + doff);
    }
  };

  ABfrag qf[4];
  {
    const u16* qp = Qb + (size_t)(qbase + wv * 16 + rr) * 128;
#pragma unroll
    for (int kk = 0; kk < 4; ++kk)
#pragma unroll
      for (int j = 0; j < 4; ++j) {
        qf[kk].u[j]     = *(const u32*)(qp + 32 * kk + half * 8 + 2 * j);
        qf[kk].u[4 + j] = *(const u32*)(qp + 32 * kk + half * 8 + 16 + 2 * j);
      }
  }
  v8f o[8];
  for (int c = 0; c < 8; ++c)
    for (int r = 0; r < 8; ++r) o[c][r] = 0.0f;
  float mi[8], li[8];
  for (int r = 0; r < 8; ++r) { mi[r] = -1e30f; li[r] = 0.0f; }

  const int nkb = (qbase >> 6) + 2;               // causal: keys < qbase+128
  prefetch(0, 0);
  for (int kb = 0; kb < nkb; ++kb) {
    const int cur = kb & 1;
    if (kb + 1 < nkb) {
      // Buffer cur^1 was released by the barrier at the end of iter kb-1.
      prefetch(kb + 1, cur ^ 1);
      // 8 newest ops (next block) may stay outstanding; current block done.
      asm volatile("s_wait_asynccnt 0x8" ::: "memory");
    } else {
      asm volatile("s_wait_asynccnt 0x0" ::: "memory");
    }
    __syncthreads();                              // all waves' DMAs visible

    // S = Q K^T : 16 x 64 per wave (4 col tiles, 4 k-steps over d)
    v8f s[4];
#pragma unroll
    for (int ct = 0; ct < 4; ++ct) {
      for (int r = 0; r < 8; ++r) s[ct][r] = 0.0f;
#pragma unroll
      for (int kk = 0; kk < 4; ++kk) {
        ABfrag bfr;
        const u16* p = &Ks[cur][ct * 16 + rr][32 * kk];
#pragma unroll
        for (int q = 0; q < 8; ++q) bfr.u[q] = *(const u32*)(p + half * 8 + (q >> 2) * 16 + 2 * (q & 3));
        s[ct] = wmma_bf16(qf[kk].v, bfr.v, s[ct]);
      }
    }

    // causal mask + online softmax (per-row stats within each 16-lane half)
    const int jb = kb * 64;
#pragma unroll
    for (int r = 0; r < 8; ++r) {
      const int irow = qbase + wv * 16 + half * 8 + r;
#pragma unroll
      for (int ct = 0; ct < 4; ++ct)
        if (jb + ct * 16 + rr > irow) s[ct][r] = -1e30f;
      float rv = fmaxf(fmaxf(s[0][r], s[1][r]), fmaxf(s[2][r], s[3][r]));
#pragma unroll
      for (int off = 1; off < 16; off <<= 1)
        rv = fmaxf(rv, __shfl_xor(rv, off, 32));
      float mn = fmaxf(mi[r], rv);
      float al = __expf(mi[r] - mn);
      float rs = 0.0f;
#pragma unroll
      for (int ct = 0; ct < 4; ++ct) {
        float pv = __expf(s[ct][r] - mn);
        s[ct][r] = pv;
        rs += pv;
      }
#pragma unroll
      for (int off = 1; off < 16; off <<= 1) rs += __shfl_xor(rs, off, 32);
      li[r] = li[r] * al + rs;
      mi[r] = mn;
#pragma unroll
      for (int c2 = 0; c2 < 8; ++c2) o[c2][r] *= al;
    }

    // P: C-layout -> LDS -> A-layout (per-wave buffer, wave-ordered LDS)
#pragma unroll
    for (int r = 0; r < 8; ++r)
#pragma unroll
      for (int ct = 0; ct < 4; ++ct)
        Ps[wv][half * 8 + r][ct * 16 + rr] = f2b(s[ct][r]);

    ABfrag pa[2];
#pragma unroll
    for (int kk2 = 0; kk2 < 2; ++kk2) {
      const u16* p = &Ps[wv][rr][32 * kk2];
#pragma unroll
      for (int j = 0; j < 4; ++j) {
        pa[kk2].u[j]     = *(const u32*)(p + half * 8 + 2 * j);
        pa[kk2].u[4 + j] = *(const u32*)(p + half * 8 + 16 + 2 * j);
      }
    }
    // O += P V : 16 x 128 per wave (8 col tiles, 2 k-steps over keys).
    // V is row-major in LDS; pack key-pairs (stride 128) per fragment dword.
#pragma unroll
    for (int c2 = 0; c2 < 8; ++c2)
#pragma unroll
      for (int kk2 = 0; kk2 < 2; ++kk2) {
        ABfrag vbf;
        const u16* p0 = &Vs[cur][32 * kk2 + half * 8][c2 * 16 + rr];
#pragma unroll
        for (int q = 0; q < 8; ++q)
          vbf.u[q] = (u32)p0[((q >> 2) * 16 + 2 * (q & 3)) * 128] | ((u32)p0[((q >> 2) * 16 + 2 * (q & 3) + 1) * 128] << 16);
        o[c2] = wmma_bf16(pa[kk2].v, vbf.v, o[c2]);
      }
    __syncthreads();                              // release buffer `cur`
  }

  const int b = bh >> 3, h = bh & 7;
#pragma unroll
  for (int c2 = 0; c2 < 8; ++c2)
#pragma unroll
    for (int r = 0; r < 8; ++r) {
      float val = o[c2][r] / li[r];
      int t = qbase + wv * 16 + half * 8 + r;
      int d = c2 * 16 + rr;
      O[((size_t)(b * 2048 + t)) * 1024 + h * 128 + d] = f2b(val);
    }
}

// ---------------------------------------------------------------------------
extern "C" void kernel_launch(void* const* d_in, const int* in_sizes, int n_in,
                              void* d_out, int out_size, void* d_ws, size_t ws_size,
                              hipStream_t stream) {
  (void)in_sizes; (void)n_in; (void)out_size; (void)ws_size;
  const float* x  = (const float*)d_in[0];
  const float* wq = (const float*)d_in[1];
  const float* wk = (const float*)d_in[2];
  const float* wv = (const float*)d_in[3];
  const float* wo = (const float*)d_in[4];

  const size_t NTOK = (size_t)8192 * 1024;        // B*T x H*D elements
  u16* qb = (u16*)d_ws;                           // 16 MB bf16 [B,H,T,D]
  u16* kb = qb + NTOK;                            // 16 MB
  u16* vb = kb + NTOK;                            // 16 MB
  u16* ob = vb + NTOK;                            // 16 MB bf16 [B,T,H*D]

  dim3 blk(256);
  dim3 gg(8, 64);                                 // N tiles x M tiles
  gemm_kernel<0, 0><<<gg, blk, 0, stream>>>(x, wq, qb);
  gemm_kernel<0, 0><<<gg, blk, 0, stream>>>(x, wk, kb);
  gemm_kernel<0, 0><<<gg, blk, 0, stream>>>(x, wv, vb);

  const int total = 4 * 8 * 2048 * 96;
  rope_kernel<<<dim3((total + 255) / 256), blk, 0, stream>>>(qb, kb);

  flash_kernel<<<dim3(16, 32), blk, 0, stream>>>(qb, kb, vb, ob);

  gemm_kernel<1, 1><<<gg, blk, 0, stream>>>(ob, wo, (float*)d_out);
}
